// VaeRNNDecoder_56556129354251
// MI455X (gfx1250) — matrix operand order repaired and checked by reading, not compiled, not run on hardware
//
#include <hip/hip_runtime.h>

// ---------------------------------------------------------------------------
// VAE LSTM autoregressive decoder for MI455X (gfx1250, wave32, WMMA).
//   B=1024, IN=128, H=192 (4H=768 gates), O_CH=128, O_LEN=512 steps.
// 32 workgroups x 256 threads (8 wave32 = 2 waves per SIMD32 for latency
// hiding). Each WG owns 32 batch rows as two independent 16-row groups;
// within a group, wave wv owns gate column tiles j=3wv..3wv+2 and output
// tiles t=2wv..2wv+1. f16 WMMA (V_WMMA_F32_16X16X32_F16) with fp32
// accumulators; cell state c stays fp32 in VGPRs; [y|h] recurrent
// activations double-buffered in LDS per group. Weights pre-packed to f16
// B-fragments in d_ws (L2-resident, 528 KB), 4-gate groups contiguous so
// each step streams sequential 4KB bursts from L2. Fragment pointers are
// laundered (asm) in address_space(1) so each step re-issues coalesced
// global_load_b128 instead of hoisting + scratch-spilling. Output stores
// are non-temporal to keep L2 for the weight stream.
// ---------------------------------------------------------------------------

typedef __attribute__((ext_vector_type(16))) _Float16 v16h;
typedef __attribute__((ext_vector_type(8)))  _Float16 v8h;
typedef __attribute__((ext_vector_type(8)))  float    v8f;

typedef const v16h __attribute__((address_space(1)))* gfrag_ptr;

#define BATCH       1024
#define IN_DIM      128
#define HID         192
#define OCH         128
#define OLEN        512
#define NKC         10       // (IN+HID)/32 gate GEMM k-chunks
#define NKH         6        // HID/32 projection k-chunks
#define XH_STRIDE   328      // 320 + 8 pad (16B-aligned rows, spreads banks)
#define ROWS_PER_WG 32       // two independent 16-row groups per WG
#define THREADS     256      // 8 wave32 -> 2 waves per SIMD32

#define GATE_FRAG_ELEMS (12 * 4 * NKC * 512)   // 480 frags * 512 f16 = 245760
#define WL_FRAG_ELEMS   (8 * NKH * 512)        //  48 frags * 512 f16 =  24576

#if __has_builtin(__builtin_amdgcn_tanhf)
// CDNA5 V_TANH_F32 transcendental (TRANS32)
__device__ __forceinline__ float ftanh(float x) { return __builtin_amdgcn_tanhf(x); }
#else
__device__ __forceinline__ float ftanh(float x) { return 2.0f / (1.0f + __expf(-2.0f * x)) - 1.0f; }
#endif
__device__ __forceinline__ float fsig(float x)  { return __builtin_fmaf(0.5f, ftanh(0.5f * x), 0.5f); }

// ---------------------------------------------------------------------------
// Pack fp32 weights into f16 WMMA B-fragments (ISA 7.12.2 layout):
//   B tile is [K=32, N=16]; lane l holds N = l%16, K = (l<16?0:8)+{0..7,16..23},
//   j16 = 0..15 f16 (matches A). Fragment = 32 lanes * 32 B, lane-contiguous.
// Gate fragment id: ((j*NKC + kc)*4 + g)  -> the 4 gates used by one WMMA
// group are 4KB-contiguous. Wl fragment id: (t*NKH + kc).
// ---------------------------------------------------------------------------
__global__ void pack_weights(const float* __restrict__ Wih,
                             const float* __restrict__ Whh,
                             const float* __restrict__ Wl,
                             _Float16* __restrict__ gateFrag,
                             _Float16* __restrict__ wlFrag) {
  int tid = blockIdx.x * blockDim.x + threadIdx.x;
  if (tid < GATE_FRAG_ELEMS) {
    int e    = tid;
    int frag = e >> 9;           // 512 f16 per fragment
    int r    = e & 511;
    int lane = r >> 4;
    int j16  = r & 15;
    int g    = frag & 3;         // gate: 0=i 1=f 2=g 3=o (PyTorch order)
    int rem  = frag >> 2;
    int kc   = rem % NKC;
    int j    = rem / NKC;        // column tile within HID
    int n    = g * HID + j * 16 + (lane & 15);
    int k    = kc * 32 + ((lane >> 4) << 3) + (j16 & 7) + ((j16 >> 3) << 4);
    float v  = (k < IN_DIM) ? Wih[n * IN_DIM + k] : Whh[n * HID + (k - IN_DIM)];
    gateFrag[e] = (_Float16)v;
  } else {
    int e = tid - GATE_FRAG_ELEMS;
    if (e < WL_FRAG_ELEMS) {
      int frag = e >> 9;
      int r    = e & 511;
      int lane = r >> 4;
      int j16  = r & 15;
      int kc   = frag % NKH;
      int t    = frag / NKH;
      int n    = t * 16 + (lane & 15);
      int k    = kc * 32 + ((lane >> 4) << 3) + (j16 & 7) + ((j16 >> 3) << 4);
      wlFrag[e] = (_Float16)Wl[n * HID + k];
    }
  }
}

// ---------------------------------------------------------------------------
// Main recurrence kernel. One WG = 32 batch rows = 2 groups x (16 rows, 4 waves).
// ---------------------------------------------------------------------------
__global__ __launch_bounds__(THREADS) void lstm_decode(
    const float* __restrict__ z,
    const float* __restrict__ bih,
    const float* __restrict__ bhh,
    const float* __restrict__ bl,
    const _Float16* __restrict__ gateFrag,
    const _Float16* __restrict__ wlFrag,
    float* __restrict__ out) {
  // per-group double-buffered [y(128) | h(192)] f16 activations
  __shared__ __align__(16) _Float16 xh[2][2][16 * XH_STRIDE];

  const int tid   = threadIdx.x;
  const int lane  = tid & 31;
  const int group = tid >> 7;      // which 16-row M-tile (0/1)
  const int gtid  = tid & 127;     // thread id within group
  const int wv    = (tid >> 5) & 3; // wave index within group
  const int nloc  = lane & 15;     // column within a 16-wide tile (B/C/D layout)
  const int hi    = lane >> 4;     // lane half
  const int mD    = hi * 8;        // C/D layout row offset for lanes 16..31
  const int wgRow = blockIdx.x * ROWS_PER_WG + group * 16;

  // --- init: xh[group][0] = [ z | h0=0 ] ; c = 0 -------------------------
  for (int idx = gtid; idx < 16 * XH_STRIDE; idx += 128) {
    int row = idx / XH_STRIDE;
    int col = idx - row * XH_STRIDE;
    float v = (col < IN_DIM) ? z[(wgRow + row) * IN_DIM + col] : 0.0f;
    xh[group][0][idx] = (_Float16)v;
  }

  // hoist the 14 per-lane bias scalars into registers (loop-invariant)
  float bsum[3][4];
#pragma unroll
  for (int jj = 0; jj < 3; ++jj)
#pragma unroll
    for (int g = 0; g < 4; ++g) {
      const int bidx = g * HID + (wv * 3 + jj) * 16 + nloc;
      bsum[jj][g] = bih[bidx] + bhh[bidx];
    }
  float blv[2];
#pragma unroll
  for (int tt = 0; tt < 2; ++tt) blv[tt] = bl[(wv * 2 + tt) * 16 + nloc];

  v8f cst[3];
#pragma unroll
  for (int jj = 0; jj < 3; ++jj)
#pragma unroll
    for (int r = 0; r < 8; ++r) cst[jj][r] = 0.0f;
  __syncthreads();

#pragma unroll 1
  for (int s = 0; s < OLEN; ++s) {
    const _Float16* rd = xh[group][s & 1];        // [y_{s-1} | h_{s-1}] (or [z|0])
    _Float16*       wr = xh[group][(s & 1) ^ 1];  // [y_s     | h_s    ]

    // ===== phase A: gates -> c,h for this wave's 3 column tiles ==========
#pragma unroll
    for (int jj = 0; jj < 3; ++jj) {
      const int j = wv * 3 + jj;
      // launder the fragment pointer in GLOBAL address space: breaks
      // loop-invariance (no hoist/spill) and lowers to global_load_b128.
      gfrag_ptr gfv = (gfrag_ptr)(unsigned long long)gateFrag;
      asm volatile("" : "+s"(gfv));

      v8f acc[4];
#pragma unroll
      for (int g = 0; g < 4; ++g)
#pragma unroll
        for (int r = 0; r < 8; ++r) acc[g][r] = bsum[jj][g];

#pragma unroll
      for (int kc = 0; kc < NKC; ++kc) {
        // A-fragment (16-bit A 16x32 layout): lane half hi holds K = hi*8 + {0..7,16..23}
        const _Float16* ap = rd + nloc * XH_STRIDE + kc * 32 + hi * 8;
        v8h alo = *(const v8h*)ap;          // ds_load_b128
        v8h ahi = *(const v8h*)(ap + 16);   // ds_load_b128
        v16h a = __builtin_shufflevector(alo, ahi,
                 0, 1, 2, 3, 4, 5, 6, 7, 8, 9, 10, 11, 12, 13, 14, 15);
#pragma unroll
        for (int g = 0; g < 4; ++g) {
          // 4 gate fragments for this (j,kc) are contiguous (4KB burst)
          v16h b = gfv[((j * NKC + kc) * 4 + g) * 32 + lane];  // 2x global_load_b128
          acc[g] = __builtin_amdgcn_wmma_f32_16x16x32_f16(
              false, a, false, b, (short)0, acc[g], false, false);
        }
      }
      // LSTM cell elementwise (fp32): c = sig(f)*c + sig(i)*tanh(g); h = sig(o)*tanh(c)
      v8f hnew;
#pragma unroll
      for (int r = 0; r < 8; ++r) {
        float ig = fsig(acc[0][r]);
        float fg = fsig(acc[1][r]);
        float gg = ftanh(acc[2][r]);
        float og = fsig(acc[3][r]);
        float cn = fg * cst[jj][r] + ig * gg;
        cst[jj][r] = cn;
        hnew[r] = og * ftanh(cn);
      }
#pragma unroll
      for (int r = 0; r < 8; ++r)
        wr[(r + mD) * XH_STRIDE + IN_DIM + j * 16 + nloc] = (_Float16)hnew[r];
    }
    __syncthreads();   // all of h_s visible in wr (both groups in lockstep)

    // ===== phase B: y_s = h_s @ Wl^T + bl (2 output tiles per wave) ======
    {
      gfrag_ptr wfv = (gfrag_ptr)(unsigned long long)wlFrag;
      asm volatile("" : "+s"(wfv));
#pragma unroll
      for (int tt = 0; tt < 2; ++tt) {
        const int t = wv * 2 + tt;
        v8f accy;
#pragma unroll
        for (int r = 0; r < 8; ++r) accy[r] = blv[tt];
#pragma unroll
        for (int kc = 0; kc < NKH; ++kc) {
          const _Float16* ap = wr + nloc * XH_STRIDE + IN_DIM + kc * 32 + hi * 8;
          v8h alo = *(const v8h*)ap;
          v8h ahi = *(const v8h*)(ap + 16);
          v16h a = __builtin_shufflevector(alo, ahi,
                   0, 1, 2, 3, 4, 5, 6, 7, 8, 9, 10, 11, 12, 13, 14, 15);
          v16h b = wfv[(t * NKH + kc) * 32 + lane];
          accy = __builtin_amdgcn_wmma_f32_16x16x32_f16(
              false, a, false, b, (short)0, accy, false, false);
        }
#pragma unroll
        for (int r = 0; r < 8; ++r) {
          const float yv = accy[r];
          const long row = (long)(wgRow + r + mD);
          // write-once output: non-temporal store keeps L2 for the weight stream
          __builtin_nontemporal_store(yv, &out[(row * OLEN + s) * OCH + t * 16 + nloc]);
          wr[(r + mD) * XH_STRIDE + t * 16 + nloc] = (_Float16)yv;   // next input
        }
      }
    }
    __syncthreads();   // y_s visible before next step reads wr
  }
}

// ---------------------------------------------------------------------------
extern "C" void kernel_launch(void* const* d_in, const int* in_sizes, int n_in,
                              void* d_out, int out_size, void* d_ws, size_t ws_size,
                              hipStream_t stream) {
  const float* z   = (const float*)d_in[0];
  const float* Wih = (const float*)d_in[1];
  const float* Whh = (const float*)d_in[2];
  const float* bih = (const float*)d_in[3];
  const float* bhh = (const float*)d_in[4];
  const float* Wl  = (const float*)d_in[5];
  const float* bl  = (const float*)d_in[6];
  float* out = (float*)d_out;

  _Float16* gateFrag = (_Float16*)d_ws;                 // 480 KB
  _Float16* wlFrag   = gateFrag + GATE_FRAG_ELEMS;      //  48 KB

  const int packElems = GATE_FRAG_ELEMS + WL_FRAG_ELEMS;
  pack_weights<<<(packElems + 255) / 256, 256, 0, stream>>>(Wih, Whh, Wl,
                                                            gateFrag, wlFrag);
  lstm_decode<<<BATCH / ROWS_PER_WG, THREADS, 0, stream>>>(
      z, bih, bhh, bl, gateFrag, wlFrag, out);
}
